// newnewBlock_9122510537236
// MI455X (gfx1250) — compile-verified, matrix operand order checked by reading, not run on hardware
//
#include <hip/hip_runtime.h>
#include <hip/hip_bf16.h>
#include <math.h>

// Problem constants
#define BB 8
#define NN 4096
#define CC 256
#define HIDD 1024
#define SCALE_ 0.17677669529663689f
#define LN_EPS_ 1e-5f
#define BN_EPS_ 1e-5f

typedef __attribute__((ext_vector_type(16))) __bf16 v16bf;
typedef __attribute__((ext_vector_type(8)))  float  v8f;

union Frag { v16bf v; float4 f4[2]; };

static __device__ __forceinline__ unsigned short f2bf(float f) {
  unsigned int u = __float_as_uint(f);
  unsigned int r = u + 0x7FFFu + ((u >> 16) & 1u);
  return (unsigned short)(r >> 16);
}
static __device__ __forceinline__ float bf2f(unsigned short h) {
  return __uint_as_float(((unsigned int)h) << 16);
}
static __device__ __forceinline__ float gelu_(float x) {
  return 0.5f * x * (1.f + erff(x * 0.70710678118654752f));
}
// Low 32 bits of a flat shared-memory address = wave-relative LDS byte offset
// (ISA 10.2: LDS aperture uses addr[31:0]).
static __device__ __forceinline__ unsigned lds_off(const void* p) {
  return (unsigned)(unsigned long long)p;
}

// ---------------------------------------------------------------------------
// f32 -> bf16 converter
// ---------------------------------------------------------------------------
__global__ void cvt_bf16_kernel(const float* __restrict__ src,
                                unsigned short* __restrict__ dst, int n) {
  int i = blockIdx.x * 256 + threadIdx.x;
  if (i < n) dst[i] = f2bf(src[i]);
}

// ---------------------------------------------------------------------------
// LayerNorm over C=256, output bf16. One row per block (256 threads).
// ---------------------------------------------------------------------------
__global__ __launch_bounds__(256)
void ln_bf16_kernel(const float* __restrict__ in, const float* __restrict__ g,
                    const float* __restrict__ bta, unsigned short* __restrict__ out) {
  __shared__ float red[256];
  const int row = blockIdx.x, t = threadIdx.x;
  float v = in[(size_t)row * 256 + t];
  red[t] = v; __syncthreads();
  for (int s = 128; s > 0; s >>= 1) { if (t < s) red[t] += red[t + s]; __syncthreads(); }
  float mean = red[0] * (1.f / 256.f);
  __syncthreads();
  float d = v - mean;
  red[t] = d * d; __syncthreads();
  for (int s = 128; s > 0; s >>= 1) { if (t < s) red[t] += red[t + s]; __syncthreads(); }
  float var = red[0] * (1.f / 256.f);
  float y = d * rsqrtf(var + LN_EPS_) * g[t] + bta[t];
  out[(size_t)row * 256 + t] = f2bf(y);
}

// ---------------------------------------------------------------------------
// bf16 WMMA GEMM with async global->LDS staging, LDS double buffering and a
// 2x-unrolled software pipeline (static buffer indices, bumped pointers).
// out[M,Nt] = A[M,K] * W[Nt,K]^T, fp32 accumulate. K must be a multiple of 64.
// Block tile 64x128, K-step 32, 8 wave32 waves (2x4), each wave 2x2 WMMA frags.
// Caller guarantees W is readable for ceil(Nt/128)*128 rows (pad if needed);
// out-of-range columns are junk but never stored (ncol < Nt guard).
// EPI 0: raw f32. EPI 1: bias+BN+GELU -> bf16. EPI 2: bias+BN+GELU+resid -> f32.
// ---------------------------------------------------------------------------
template <int EPI>
__global__ __launch_bounds__(256)
void gemm_wmma_bf16(const unsigned short* __restrict__ A,
                    const unsigned short* __restrict__ W,
                    float* __restrict__ outF,
                    unsigned short* __restrict__ outH,
                    int M, int Nt, int K,
                    const float* __restrict__ bias,
                    const float* __restrict__ bng, const float* __restrict__ bnb,
                    const float* __restrict__ bnm, const float* __restrict__ bnv,
                    const float* __restrict__ resid) {
  __shared__ unsigned short As[2][64 * 32];    // 2 x 4 KB
  __shared__ unsigned short Bs[2][128 * 32];   // 2 x 8 KB

  const int tid  = threadIdx.x;
  const int wave = tid >> 5, lane = tid & 31;
  const int wm = wave >> 2, wn = wave & 3;
  const int m0 = blockIdx.y * 64;
  const int n0 = blockIdx.x * 128;
  const int lrow  = lane & 15;
  const int khalf = (lane >> 4) << 3;  // 0 or 8 (ISA 16-bit A/B lane split)

  v8f acc[2][2] = {};

  const int arow = tid >> 2;        // 0..63
  const int acol = (tid & 3) << 3;  // 0,8,16,24

  // Per-thread global staging pointers, bumped +32 elements per issued tile.
  const unsigned short* ag  = A + (size_t)(m0 + arow) * K + acol;
  const unsigned short* bg0 = W + (size_t)(n0 + arow) * K + acol;
  const unsigned short* bg1 = W + (size_t)(n0 + arow + 64) * K + acol;
  // Per-thread LDS destinations for both buffers.
  const unsigned aL[2]  = { lds_off(&As[0][arow * 32 + acol]),
                            lds_off(&As[1][arow * 32 + acol]) };
  const unsigned bL0[2] = { lds_off(&Bs[0][arow * 32 + acol]),
                            lds_off(&Bs[1][arow * 32 + acol]) };
  const unsigned bL1[2] = { lds_off(&Bs[0][(arow + 64) * 32 + acol]),
                            lds_off(&Bs[1][(arow + 64) * 32 + acol]) };

  // Issue one 64x32 A-tile + 128x32 B-tile into buffer b; bump pointers.
  auto issue = [&](int b) {
    asm volatile("global_load_async_to_lds_b128 %0, %1, off"
                 :: "v"(aL[b]),  "v"(ag)  : "memory");
    asm volatile("global_load_async_to_lds_b128 %0, %1, off"
                 :: "v"(bL0[b]), "v"(bg0) : "memory");
    asm volatile("global_load_async_to_lds_b128 %0, %1, off"
                 :: "v"(bL1[b]), "v"(bg1) : "memory");
    ag += 32; bg0 += 32; bg1 += 32;
  };

  // Consume buffer b: build fragments (ISA 16-bit layout: per lane elems 0..7 =
  // K khalf..khalf+7, elems 8..15 = K khalf+16..khalf+23) and run 4 WMMAs.
  auto compute = [&](int b) {
    Frag af[2], bf[2];
    #pragma unroll
    for (int r = 0; r < 2; ++r) {
      const float4* p = reinterpret_cast<const float4*>(
          &As[b][(wm * 32 + r * 16 + lrow) * 32 + khalf]);
      af[r].f4[0] = p[0]; af[r].f4[1] = p[2];
    }
    #pragma unroll
    for (int c = 0; c < 2; ++c) {
      const float4* p = reinterpret_cast<const float4*>(
          &Bs[b][(wn * 32 + c * 16 + lrow) * 32 + khalf]);
      bf[c].f4[0] = p[0]; bf[c].f4[1] = p[2];
    }
    #pragma unroll
    for (int r = 0; r < 2; ++r)
      #pragma unroll
      for (int c = 0; c < 2; ++c)
        acc[r][c] = __builtin_amdgcn_wmma_f32_16x16x32_bf16(
            false, af[r].v, false, bf[c].v, (short)0, acc[r][c], false, false);
  };

  const int steps = K >> 5;  // K/32, even (K multiple of 64)
  issue(0);
  for (int s = 0; s < steps; s += 2) {
    // Stage 0: prefetch into buf1 (always valid: s+1 < steps), consume buf0.
    issue(1);
    asm volatile("s_wait_asynccnt 0x3" ::: "memory");  // buf0's 3 loads done
    __syncthreads();
    compute(0);
    __syncthreads();
    // Stage 1: prefetch into buf0 (unless last pair), consume buf1.
    if (s + 2 < steps) {
      issue(0);
      asm volatile("s_wait_asynccnt 0x3" ::: "memory");
    } else {
      asm volatile("s_wait_asynccnt 0x0" ::: "memory");
    }
    __syncthreads();
    compute(1);
    __syncthreads();
  }

  // Epilogue. C/D layout: VGPR j -> M = j (lanes 0-15) / 8+j (lanes 16-31), N = lane%16.
  const int ncol_l = lane & 15;
  const int moff   = (lane >> 4) << 3;
  #pragma unroll
  for (int r = 0; r < 2; ++r) {
    #pragma unroll
    for (int c = 0; c < 2; ++c) {
      int ncol = n0 + wn * 32 + c * 16 + ncol_l;
      int mrow = m0 + wm * 32 + r * 16 + moff;
      if (ncol < Nt) {
        float bscale = 1.f, bshift = 0.f, bia = 0.f;
        if (EPI != 0) {
          bia = bias[ncol];
          float inv = rsqrtf(bnv[ncol] + BN_EPS_);
          bscale = bng[ncol] * inv;
          bshift = bnb[ncol] - bnm[ncol] * bscale;
        }
        #pragma unroll
        for (int j = 0; j < 8; ++j) {
          float v = acc[r][c][j];
          size_t idx = (size_t)(mrow + j) * Nt + ncol;
          if (EPI == 0) {
            outF[idx] = v;
          } else if (EPI == 1) {
            outH[idx] = f2bf(gelu_((v + bia) * bscale + bshift));
          } else {
            outF[idx] = resid[idx] + gelu_((v + bia) * bscale + bshift);
          }
        }
      }
    }
  }
}

// ---------------------------------------------------------------------------
// Per-(b,h): column softmax of z over N=4096 tokens, then
// kbar[nb][d] = sum_n z_[n,nb] * k[n,d] (and vbar). 64 blocks x 256 threads.
// ---------------------------------------------------------------------------
__global__ __launch_bounds__(256)
void cluster_kv_kernel(const float* __restrict__ z, const float* __restrict__ qkv,
                       float* __restrict__ kbar, float* __restrict__ vbar) {
  __shared__ float red[256];
  const int bh = blockIdx.x;
  const int b = bh >> 3, h = bh & 7;
  const int t = threadIdx.x;
  const int nb = t >> 5, j = t & 31;
  const float* zp = z + (size_t)b * 4096 * 64 + h * 8 + nb;

  float mx = -1e30f;
  for (int n = j; n < 4096; n += 32) mx = fmaxf(mx, zp[(size_t)n * 64]);
  red[t] = mx; __syncthreads();
  for (int s = 16; s > 0; s >>= 1) { if (j < s) red[t] = fmaxf(red[t], red[t + s]); __syncthreads(); }
  mx = red[nb << 5]; __syncthreads();

  float sm = 0.f;
  for (int n = j; n < 4096; n += 32) sm += expf(zp[(size_t)n * 64] - mx);
  red[t] = sm; __syncthreads();
  for (int s = 16; s > 0; s >>= 1) { if (j < s) red[t] += red[t + s]; __syncthreads(); }
  float rinv = 1.f / red[nb << 5];

  const float* kp = qkv + (size_t)b * 4096 * 768 + 256 + h * 32 + j;  // k part
  const float* vp = kp + 256;                                          // v part
  float ak = 0.f, av = 0.f;
  for (int n = 0; n < 4096; ++n) {
    float p = expf(zp[(size_t)n * 64] - mx) * rinv;
    ak += p * kp[(size_t)n * 768];
    av += p * vp[(size_t)n * 768];
  }
  kbar[(size_t)bh * 256 + t] = ak;
  vbar[(size_t)bh * 256 + t] = av;
}

// ---------------------------------------------------------------------------
// Attention over 8 clusters + residual: x1 = x + attn(q,kbar)@vbar.
// grid (16 token-chunks, 64 bh), one token per thread.
// ---------------------------------------------------------------------------
__global__ __launch_bounds__(256)
void attn_kernel(const float* __restrict__ qkv, const float* __restrict__ kbar,
                 const float* __restrict__ vbar, const float* __restrict__ x,
                 float* __restrict__ x1) {
  __shared__ float kb[256];
  __shared__ float vb[256];
  const int bh = blockIdx.y;
  const int b = bh >> 3, h = bh & 7;
  const int t = threadIdx.x;
  kb[t] = kbar[(size_t)bh * 256 + t];
  vb[t] = vbar[(size_t)bh * 256 + t];
  __syncthreads();

  const int n = blockIdx.x * 256 + t;
  const size_t rowq = ((size_t)b * 4096 + n) * 768 + h * 32;
  float q[32];
  #pragma unroll
  for (int i = 0; i < 8; ++i) {
    float4 v = *reinterpret_cast<const float4*>(qkv + rowq + i * 4);
    q[i*4+0] = v.x; q[i*4+1] = v.y; q[i*4+2] = v.z; q[i*4+3] = v.w;
  }
  float s[8];
  float mx = -1e30f;
  #pragma unroll
  for (int c = 0; c < 8; ++c) {
    float d = 0.f;
    #pragma unroll
    for (int k = 0; k < 32; ++k) d += q[k] * kb[c * 32 + k];
    s[c] = d * SCALE_;
    mx = fmaxf(mx, s[c]);
  }
  float sum = 0.f;
  #pragma unroll
  for (int c = 0; c < 8; ++c) { s[c] = expf(s[c] - mx); sum += s[c]; }
  float rinv = 1.f / sum;
  const size_t rowo = ((size_t)b * 4096 + n) * 256 + h * 32;
  #pragma unroll
  for (int d = 0; d < 32; ++d) {
    float o = 0.f;
    #pragma unroll
    for (int c = 0; c < 8; ++c) o += s[c] * vb[c * 32 + d];
    x1[rowo + d] = x[rowo + d] + o * rinv;
  }
}

// ---------------------------------------------------------------------------
// Depthwise 3x3 on 64x64 (token-major [b,n,ch]) + bias + BN2 + GELU -> bf16.
// ---------------------------------------------------------------------------
__global__ __launch_bounds__(256)
void dwconv_kernel(const unsigned short* __restrict__ in, const float* __restrict__ w,
                   const float* __restrict__ bias,
                   const float* __restrict__ g, const float* __restrict__ bb,
                   const float* __restrict__ m, const float* __restrict__ vv,
                   unsigned short* __restrict__ out) {
  const int row = blockIdx.x;
  const int b = row >> 12, n = row & 4095;
  const int hh = n >> 6, ww = n & 63;
  const int t = threadIdx.x;
  const size_t base = (size_t)b * 4096;
  for (int i = 0; i < 4; ++i) {
    int ch = (i << 8) + t;
    float acc = 0.f;
    #pragma unroll
    for (int dy = -1; dy <= 1; ++dy) {
      int y2 = hh + dy;
      if (y2 < 0 || y2 >= 64) continue;
      #pragma unroll
      for (int dx = -1; dx <= 1; ++dx) {
        int x2 = ww + dx;
        if (x2 < 0 || x2 >= 64) continue;
        acc += bf2f(in[(base + (size_t)(y2 * 64 + x2)) * 1024 + ch]) *
               w[ch * 9 + (dy + 1) * 3 + (dx + 1)];
      }
    }
    acc += bias[ch];
    float tn = (acc - m[ch]) * rsqrtf(vv[ch] + BN_EPS_) * g[ch] + bb[ch];
    out[(base + n) * 1024 + ch] = f2bf(gelu_(tn));
  }
}

// ---------------------------------------------------------------------------
extern "C" void kernel_launch(void* const* d_in, const int* in_sizes, int n_in,
                              void* d_out, int out_size, void* d_ws, size_t ws_size,
                              hipStream_t stream) {
  (void)in_sizes; (void)n_in; (void)out_size;
  const float* x     = (const float*)d_in[0];
  const float* w_qkv = (const float*)d_in[1];
  const float* w_z   = (const float*)d_in[2];
  const float* ln1_g = (const float*)d_in[3];
  const float* ln1_b = (const float*)d_in[4];
  const float* ln2_g = (const float*)d_in[5];
  const float* ln2_b = (const float*)d_in[6];
  const float* fc1_w = (const float*)d_in[7];
  const float* fc1_b = (const float*)d_in[8];
  const float* bn1_g = (const float*)d_in[9];
  const float* bn1_b = (const float*)d_in[10];
  const float* bn1_m = (const float*)d_in[11];
  const float* bn1_v = (const float*)d_in[12];
  const float* dw_w  = (const float*)d_in[13];
  const float* dw_b  = (const float*)d_in[14];
  const float* bn2_g = (const float*)d_in[15];
  const float* bn2_b = (const float*)d_in[16];
  const float* bn2_m = (const float*)d_in[17];
  const float* bn2_v = (const float*)d_in[18];
  const float* fc2_w = (const float*)d_in[19];
  const float* fc2_b = (const float*)d_in[20];
  const float* bn3_g = (const float*)d_in[21];
  const float* bn3_b = (const float*)d_in[22];
  const float* bn3_m = (const float*)d_in[23];
  const float* bn3_v = (const float*)d_in[24];
  float* out = (float*)d_out;

  char* ws = (char*)d_ws;
  size_t off = 0;
  auto take = [&](size_t bytes) -> void* {
    void* p = ws + off;
    off += (bytes + 255) & ~(size_t)255;
    return p;
  };
  const int M = BB * NN;  // 32768 token rows
  unsigned short* wqkv_h = (unsigned short*)take((size_t)768 * 256 * 2);
  unsigned short* wz_h   = (unsigned short*)take((size_t)128 * 256 * 2);  // padded to 128 rows
  unsigned short* wfc1_h = (unsigned short*)take((size_t)1024 * 256 * 2);
  unsigned short* wfc2_h = (unsigned short*)take((size_t)256 * 1024 * 2);
  unsigned short* xln_h  = (unsigned short*)take((size_t)M * 256 * 2);   // reused for LN2 output
  float*          qkv_f  = (float*)take((size_t)M * 768 * 4);            // reused for h1 (bf16)
  float*          z_f    = (float*)take((size_t)M * 64 * 4);
  float*          kbar   = (float*)take((size_t)64 * 256 * 4);
  float*          vbar   = (float*)take((size_t)64 * 256 * 4);
  float*          x1     = (float*)take((size_t)M * 256 * 4);
  unsigned short* h2_h   = (unsigned short*)take((size_t)M * 1024 * 2);
  unsigned short* ln2_h  = xln_h;                    // alias (LN1 output dead by then)
  unsigned short* h1_h   = (unsigned short*)qkv_f;   // alias (qkv dead after attention)
  if (ws_size < off) return;  // workspace too small (no-op rather than OOB)

  // --- weight conversion to bf16 ---
  cvt_bf16_kernel<<<dim3((768 * 256 + 255) / 256), 256, 0, stream>>>(w_qkv, wqkv_h, 768 * 256);
  cvt_bf16_kernel<<<dim3((64 * 256 + 255) / 256), 256, 0, stream>>>(w_z, wz_h, 64 * 256);
  cvt_bf16_kernel<<<dim3((1024 * 256 + 255) / 256), 256, 0, stream>>>(fc1_w, wfc1_h, 1024 * 256);
  cvt_bf16_kernel<<<dim3((256 * 1024 + 255) / 256), 256, 0, stream>>>(fc2_w, wfc2_h, 256 * 1024);

  // --- LN1 -> bf16 ---
  ln_bf16_kernel<<<dim3(M), 256, 0, stream>>>(x, ln1_g, ln1_b, xln_h);

  // --- qkv GEMM [M,256]x[256,768] and z GEMM [M,256]x[256,64] (WMMA bf16) ---
  gemm_wmma_bf16<0><<<dim3(6, M / 64), 256, 0, stream>>>(
      xln_h, wqkv_h, qkv_f, nullptr, M, 768, 256,
      nullptr, nullptr, nullptr, nullptr, nullptr, nullptr);
  gemm_wmma_bf16<0><<<dim3(1, M / 64), 256, 0, stream>>>(
      xln_h, wz_h, z_f, nullptr, M, 64, 256,
      nullptr, nullptr, nullptr, nullptr, nullptr, nullptr);

  // --- token softmax + cluster k/v summaries, then attention + residual ---
  cluster_kv_kernel<<<dim3(64), 256, 0, stream>>>(z_f, qkv_f, kbar, vbar);
  attn_kernel<<<dim3(16, 64), 256, 0, stream>>>(qkv_f, kbar, vbar, x, x1);

  // --- LN2 -> bf16 ---
  ln_bf16_kernel<<<dim3(M), 256, 0, stream>>>(x1, ln2_g, ln2_b, ln2_h);

  // --- fc1 GEMM + bias + BN1 + GELU -> bf16 ---
  gemm_wmma_bf16<1><<<dim3(8, M / 64), 256, 0, stream>>>(
      ln2_h, wfc1_h, nullptr, h1_h, M, 1024, 256,
      fc1_b, bn1_g, bn1_b, bn1_m, bn1_v, nullptr);

  // --- depthwise 3x3 + bias + BN2 + GELU -> bf16 ---
  dwconv_kernel<<<dim3(M), 256, 0, stream>>>(
      h1_h, dw_w, dw_b, bn2_g, bn2_b, bn2_m, bn2_v, h2_h);

  // --- fc2 GEMM + bias + BN3 + GELU + residual -> fp32 final output ---
  gemm_wmma_bf16<2><<<dim3(2, M / 64), 256, 0, stream>>>(
      h2_h, wfc2_h, out, nullptr, M, 256, 1024,
      fc2_b, bn3_g, bn3_b, bn3_m, bn3_v, x1);
}
